// ReprogrammingLayer_MLA_35622458753687
// MI455X (gfx1250) — matrix operand order repaired and checked by reading, not compiled, not run on hardware
//
#include <hip/hip_runtime.h>
#include <hip/hip_bf16.h>

// ---------------------------------------------------------------------------
// CDNA5 (gfx1250) MLA pipeline: bf16 WMMA GEMMs + flash attention with
// async-to-LDS K staging.
// ---------------------------------------------------------------------------

typedef __attribute__((ext_vector_type(16))) __bf16 v16bf;
typedef __attribute__((ext_vector_type(8)))  __bf16 bf16x8;
typedef __attribute__((ext_vector_type(4)))  __bf16 bf16x4;
typedef __attribute__((ext_vector_type(8)))  float  v8f;

#define LOG2E_F 1.4426950408889634f

static __device__ inline v8f zero8() {
  v8f z;
#pragma unroll
  for (int i = 0; i < 8; ++i) z[i] = 0.0f;
  return z;
}

static __device__ inline bf16x8 zerobf8() {
  bf16x8 z;
#pragma unroll
  for (int i = 0; i < 8; ++i) z[i] = (__bf16)0.0f;
  return z;
}

static __device__ inline v16bf cat8(bf16x8 lo, bf16x8 hi) {
  return __builtin_shufflevector(lo, hi, 0, 1, 2, 3, 4, 5, 6, 7,
                                 8, 9, 10, 11, 12, 13, 14, 15);
}

// A-fragment (16x32, M x K): lane = row (mod 16); lanes 0-15 hold K 0..7 and
// 16..23, lanes 16-31 hold K 8..15 and 24..31 (ISA 7.12.2).
static __device__ inline v16bf load_a_frag(const __bf16* base, int stride, int lane) {
  const int row  = lane & 15;
  const int koff = (lane >> 4) << 3;        // 0 or 8
  const __bf16* p = base + row * stride + koff;
  bf16x8 lo = *(const bf16x8*)(p);
  bf16x8 hi = *(const bf16x8*)(p + 16);
  return cat8(lo, hi);
}

// B-fragment (32x16, K x N) read from a tile stored [n][k]:
// lanes 0-15: n = lane, K 0..7 and 16..23 ; lanes 16-31: n = lane-16, K 8..15 and 24..31.
static __device__ inline v16bf load_b_frag(const __bf16* base, int stride, int lane) {
  const int n    = lane & 15;
  const int koff = (lane >> 4) << 3;        // 0 or 8
  const __bf16* p = base + n * stride + koff;
  bf16x8 lo = *(const bf16x8*)(p);
  bf16x8 hi = *(const bf16x8*)(p + 16);
  return cat8(lo, hi);
}

static __device__ inline v8f wmma_bf16(v16bf a, v16bf b, v8f c) {
  return __builtin_amdgcn_wmma_f32_16x16x32_bf16(
      /*neg_a=*/false, a, /*neg_b=*/false, b,
      /*c_mod=*/(short)0, c, /*reuse_a=*/false, /*reuse_b=*/false);
}

// ---------------------------------------------------------------------------
// Elementwise f32 -> bf16 conversion, 4-wide (n must be a multiple of 4)
// ---------------------------------------------------------------------------
__global__ void cvt_f32_bf16(const float* __restrict__ in, __bf16* __restrict__ out, long n4) {
  long i  = (long)blockIdx.x * 256 + threadIdx.x;
  long st = (long)gridDim.x * 256;
  for (; i < n4; i += st) {
    const float4 v = ((const float4*)in)[i];
    bf16x4 o;
    o[0] = (__bf16)v.x; o[1] = (__bf16)v.y; o[2] = (__bf16)v.z; o[3] = (__bf16)v.w;
    ((bf16x4*)out)[i] = o;
  }
}

// ---------------------------------------------------------------------------
// Generic GEMM: C[M,N] = A[M,K] @ W[N,K]^T (+ bias), bf16 in / f32 acc.
// Block = 256 threads (8 waves). Tile 128x128, K-chunk 32.
// Each wave computes 2x4 = 8 16x16 WMMA tiles.
// ---------------------------------------------------------------------------
template <bool OUT_BF16, bool HAS_BIAS>
__global__ __launch_bounds__(256) void gemm_xwt(
    const __bf16* __restrict__ A, const __bf16* __restrict__ W,
    const float* __restrict__ bias, void* __restrict__ Cout,
    int M, int N, int K) {
  __shared__ __bf16 As[128][40];   // [m][k], padded stride
  __shared__ __bf16 Bs[128][40];   // [n][k], matches W layout

  const int tid  = threadIdx.x;
  const int lane = tid & 31;
  const int w    = tid >> 5;
  const int wm   = w >> 1;            // 0..3
  const int wn   = w & 1;             // 0..1
  const int m0   = blockIdx.y * 128;
  const int n0   = blockIdx.x * 128;
  const int lrow = tid >> 2;          // 0..63
  const int lk8  = (tid & 3) << 3;    // 0,8,16,24

  v8f acc[2][4];
#pragma unroll
  for (int i = 0; i < 2; ++i)
#pragma unroll
    for (int j = 0; j < 4; ++j) acc[i][j] = zero8();

  const bf16x8 zv = zerobf8();

  for (int k0 = 0; k0 < K; k0 += 32) {
    __syncthreads();
#pragma unroll
    for (int rr = 0; rr < 128; rr += 64) {
      const int row = lrow + rr;
      const long am = (long)(m0 + row);
      bf16x8 va = (am < (long)M) ? *(const bf16x8*)(A + am * K + k0 + lk8) : zv;
      *(bf16x8*)(&As[row][lk8]) = va;
      bf16x8 vb = *(const bf16x8*)(W + (long)(n0 + row) * K + k0 + lk8);
      *(bf16x8*)(&Bs[row][lk8]) = vb;
      // prefetch next K-chunk into cache (global_prefetch_b8)
      if (k0 + 32 < K) {
        if (am < (long)M) __builtin_prefetch(A + am * K + k0 + 32 + lk8, 0, 0);
        __builtin_prefetch(W + (long)(n0 + row) * K + k0 + 32 + lk8, 0, 0);
      }
    }
    __syncthreads();

    v16bf af[2], bfr[4];
#pragma unroll
    for (int i = 0; i < 2; ++i) af[i] = load_a_frag(&As[(wm * 2 + i) * 16][0], 40, lane);
#pragma unroll
    for (int j = 0; j < 4; ++j) bfr[j] = load_b_frag(&Bs[(wn * 4 + j) * 16][0], 40, lane);
#pragma unroll
    for (int i = 0; i < 2; ++i)
#pragma unroll
      for (int j = 0; j < 4; ++j) acc[i][j] = wmma_bf16(af[i], bfr[j], acc[i][j]);
  }

  // C/D layout: col = lane%16, row = r + 8*(lane/16)
  const int crow0 = (lane >> 4) * 8;
  const int ccol  = lane & 15;
#pragma unroll
  for (int i = 0; i < 2; ++i) {
#pragma unroll
    for (int j = 0; j < 4; ++j) {
      const int coln = n0 + (wn * 4 + j) * 16 + ccol;
      const float bv = HAS_BIAS ? bias[coln] : 0.0f;
#pragma unroll
      for (int r = 0; r < 8; ++r) {
        const int rowm = m0 + (wm * 2 + i) * 16 + crow0 + r;
        if (rowm < M) {
          const float val = acc[i][j][r] + bv;
          if (OUT_BF16)
            ((__bf16*)Cout)[(long)rowm * N + coln] = (__bf16)val;
          else
            ((float*)Cout)[(long)rowm * N + coln] = val;
        }
      }
    }
  }
}

// ---------------------------------------------------------------------------
// Row layernorm (512 wide) -> bf16. One block (256 threads) per row.
// ---------------------------------------------------------------------------
__global__ __launch_bounds__(256) void ln512_bf16(
    const float* __restrict__ x, const float* __restrict__ g,
    const float* __restrict__ b, __bf16* __restrict__ out) {
  const int row = blockIdx.x;
  const int tid = threadIdx.x;
  const float* xr = x + (long)row * 512;
  const float v0 = xr[tid];
  const float v1 = xr[tid + 256];
  float s  = v0 + v1;
  float s2 = v0 * v0 + v1 * v1;
#pragma unroll
  for (int m = 1; m < 32; m <<= 1) {
    s  += __shfl_xor(s, m, 32);
    s2 += __shfl_xor(s2, m, 32);
  }
  __shared__ float psum[8], psum2[8];
  if ((tid & 31) == 0) { psum[tid >> 5] = s; psum2[tid >> 5] = s2; }
  __syncthreads();
  float ts = 0.0f, ts2 = 0.0f;
#pragma unroll
  for (int i = 0; i < 8; ++i) { ts += psum[i]; ts2 += psum2[i]; }
  const float mu   = ts * (1.0f / 512.0f);
  const float var  = ts2 * (1.0f / 512.0f) - mu * mu;
  const float rstd = rsqrtf(var + 1e-5f);
  out[(long)row * 512 + tid]       = (__bf16)((v0 - mu) * rstd * g[tid] + b[tid]);
  out[(long)row * 512 + tid + 256] = (__bf16)((v1 - mu) * rstd * g[tid + 256] + b[tid + 256]);
}

// ---------------------------------------------------------------------------
// Repack kv[S,2048] -> k[h][s][64], v[h][s][64], zero-padded to s=1024.
// ---------------------------------------------------------------------------
__global__ void repack_kv(const __bf16* __restrict__ kv,
                          __bf16* __restrict__ kbuf, __bf16* __restrict__ vbuf) {
  const int idx = blockIdx.x * 256 + threadIdx.x;   // [h(4b)][s(10b)][e(6b)]
  const int e = idx & 63;
  const int s = (idx >> 6) & 1023;
  const int h = idx >> 16;
  __bf16 kk = (__bf16)0.0f, vv = (__bf16)0.0f;
  if (s < 1000) {
    kk = kv[(long)s * 2048 + h * 128 + e];
    vv = kv[(long)s * 2048 + h * 128 + 64 + e];
  }
  kbuf[idx] = kk;
  vbuf[idx] = vv;
}

// ---------------------------------------------------------------------------
// Flash attention. Block = (b,h, 128 L rows); 8 waves x 16 rows each.
// K chunk staged via GLOBAL_LOAD_ASYNC_TO_LDS_B128 (ASYNCcnt); V chunk
// transposed through VGPRs into LDS; online softmax in log2 domain.
// Padded rows (s >= 1000) are zero in kbuf/vbuf and masked to -inf in scores,
// so chunk staging needs no guards.
// ---------------------------------------------------------------------------
__global__ __launch_bounds__(256) void mla_attn(
    const __bf16* __restrict__ q,     // [B*L][1024]  (h*64+e columns)
    const __bf16* __restrict__ kbuf,  // [16][1024][64]
    const __bf16* __restrict__ vbuf,  // [16][1024][64]
    __bf16* __restrict__ attn) {      // [B*L][1024]
  __shared__ __bf16 kt[32][72];       // [s][e]  -> B-frag layout for QK^T
  __shared__ __bf16 vt[64][40];       // [e][s]  -> B-frag layout for AV
  __shared__ __bf16 ps[8][16][40];    // per-wave P tile [row][k]

  const int tid  = threadIdx.x;
  const int lane = tid & 31;
  const int w    = tid >> 5;
  const int b    = blockIdx.x >> 4;
  const int h    = blockIdx.x & 15;
  const int l0   = blockIdx.y * 128 + w * 16;
  const int half = lane >> 4;
  const int ccol = lane & 15;
  const float SC = 0.125f * LOG2E_F;  // 1/sqrt(64) folded into log2 domain

  // Q fragments for this wave's 16 rows (K = e, 64 wide -> 2 K-steps)
  const __bf16* qbase = q + (long)(b * 1024 + l0) * 1024 + h * 64;
  v16bf qf[2];
#pragma unroll
  for (int ks = 0; ks < 2; ++ks) qf[ks] = load_a_frag(qbase + ks * 32, 1024, lane);

  v8f O[4];
#pragma unroll
  for (int et = 0; et < 4; ++et) O[et] = zero8();
  float mrow[8], lrow[8];
#pragma unroll
  for (int r = 0; r < 8; ++r) { mrow[r] = -1e30f; lrow[r] = 0.0f; }

  const int sr = tid >> 3;          // 0..31 (source row in chunk)
  const int eq = (tid & 7) << 3;    // 0..56 (e offset, 8-wide)
  // LDS byte offset of this thread's K-tile destination (low 32 bits of the
  // generic shared-aperture address == workgroup LDS offset).
  const unsigned kt_lds = (unsigned)(uintptr_t)(void*)&kt[sr][eq];

  for (int s0 = 0; s0 < 1024; s0 += 32) {
    __syncthreads();
    {
      const long base = (long)(h * 1024 + s0 + sr) * 64 + eq;
      // K tile: async DMA global -> LDS, 16B per lane (tracked by ASYNCcnt)
      const __bf16* kp = kbuf + base;
      asm volatile("global_load_async_to_lds_b128 %0, %1, off"
                   :: "v"(kt_lds), "v"(kp)
                   : "memory");
      // V tile: load row, transpose into [e][s] through VGPRs
      bf16x8 v8 = *(const bf16x8*)(vbuf + base);
#pragma unroll
      for (int ii = 0; ii < 8; ++ii) vt[eq + ii][sr] = v8[ii];
    }
    asm volatile("s_wait_asynccnt 0" ::: "memory");
    __syncthreads();

    // Preload all fragments so DS loads pipeline ahead of the matrix ops.
    v16bf bk[2][2], bv[4];
#pragma unroll
    for (int nt = 0; nt < 2; ++nt)
#pragma unroll
      for (int ks = 0; ks < 2; ++ks)
        bk[nt][ks] = load_b_frag(&kt[nt * 16][ks * 32], 72, lane);
#pragma unroll
    for (int et = 0; et < 4; ++et) bv[et] = load_b_frag(&vt[et * 16][0], 40, lane);

    // scores (16 rows x 32 source cols) = Q(16x64) . K^T
    v8f sc0 = zero8(), sc1 = zero8();
    sc0 = wmma_bf16(qf[0], bk[0][0], sc0);
    sc1 = wmma_bf16(qf[0], bk[1][0], sc1);
    sc0 = wmma_bf16(qf[1], bk[0][1], sc0);
    sc1 = wmma_bf16(qf[1], bk[1][1], sc1);

    const int sg0 = s0 + ccol;
    const int sg1 = s0 + 16 + ccol;
    float t[8];
#pragma unroll
    for (int r = 0; r < 8; ++r) {
      float x0 = sc0[r] * SC; if (sg0 >= 1000) x0 = -1e30f;
      float x1 = sc1[r] * SC; if (sg1 >= 1000) x1 = -1e30f;
      sc0[r] = x0; sc1[r] = x1;
      t[r] = fmaxf(x0, x1);
    }
    // row max across the 16 columns (within each wave32 half)
#pragma unroll
    for (int msk = 1; msk < 16; msk <<= 1)
#pragma unroll
      for (int r = 0; r < 8; ++r) t[r] = fmaxf(t[r], __shfl_xor(t[r], msk, 32));

    float al[8], rs[8];
#pragma unroll
    for (int r = 0; r < 8; ++r) {
      const float mn = fmaxf(mrow[r], t[r]);
      al[r] = __builtin_amdgcn_exp2f(mrow[r] - mn);
      mrow[r] = mn;
      const float p0 = __builtin_amdgcn_exp2f(sc0[r] - mn);
      const float p1 = __builtin_amdgcn_exp2f(sc1[r] - mn);
      rs[r] = p0 + p1;
      ps[w][half * 8 + r][ccol]      = (__bf16)p0;   // C-layout -> row-major P
      ps[w][half * 8 + r][16 + ccol] = (__bf16)p1;
    }
#pragma unroll
    for (int msk = 1; msk < 16; msk <<= 1)
#pragma unroll
      for (int r = 0; r < 8; ++r) rs[r] += __shfl_xor(rs[r], msk, 32);
#pragma unroll
    for (int r = 0; r < 8; ++r) lrow[r] = lrow[r] * al[r] + rs[r];
#pragma unroll
    for (int et = 0; et < 4; ++et)
#pragma unroll
      for (int r = 0; r < 8; ++r) O[et][r] *= al[r];

    // Same-wave LDS RAW across lanes: DS pipeline is in-order per wave, but
    // keep an explicit wait + memory clobber so the scheduler cannot hoist
    // the A-fragment load above the P stores.
    asm volatile("s_wait_dscnt 0" ::: "memory");
    v16bf pf = load_a_frag(&ps[w][0][0], 40, lane);
#pragma unroll
    for (int et = 0; et < 4; ++et) O[et] = wmma_bf16(pf, bv[et], O[et]);
  }

  float inv[8];
#pragma unroll
  for (int r = 0; r < 8; ++r) inv[r] = 1.0f / lrow[r];
#pragma unroll
  for (int et = 0; et < 4; ++et)
#pragma unroll
    for (int r = 0; r < 8; ++r) {
      const long rowg = (long)(b * 1024 + l0 + half * 8 + r);
      attn[rowg * 1024 + h * 64 + et * 16 + ccol] = (__bf16)(O[et][r] * inv[r]);
    }
}

// ---------------------------------------------------------------------------
// Host launcher
// ---------------------------------------------------------------------------
extern "C" void kernel_launch(void* const* d_in, const int* in_sizes, int n_in,
                              void* d_out, int out_size, void* d_ws, size_t ws_size,
                              hipStream_t stream) {
  (void)in_sizes; (void)n_in; (void)out_size; (void)ws_size;
  const float* te      = (const float*)d_in[0];
  const float* se      = (const float*)d_in[1];
  // d_in[2] (value_embedding) is unused by the reference
  const float* q_a_w   = (const float*)d_in[3];
  const float* q_ln_g  = (const float*)d_in[4];
  const float* q_ln_b  = (const float*)d_in[5];
  const float* q_b_w   = (const float*)d_in[6];
  const float* kv_a_w  = (const float*)d_in[7];
  const float* kv_ln_g = (const float*)d_in[8];
  const float* kv_ln_b = (const float*)d_in[9];
  const float* kv_b_w  = (const float*)d_in[10];
  const float* out_w   = (const float*)d_in[11];
  const float* out_b   = (const float*)d_in[12];

  char* p = (char*)d_ws;
  auto alloc = [&](size_t bytes) -> char* {
    char* r = p;
    p += (bytes + 255) & ~(size_t)255;
    return r;
  };
  __bf16* te_bf   = (__bf16*)alloc(4194304ull * 2);  // [4096][1024]
  __bf16* se_bf   = (__bf16*)alloc(4096000ull * 2);  // [1000][4096]
  __bf16* w_qa    = (__bf16*)alloc(524288ull * 2);   // [512][1024]
  __bf16* w_qb    = (__bf16*)alloc(524288ull * 2);   // [1024][512]
  __bf16* w_kva   = (__bf16*)alloc(2097152ull * 2);  // [512][4096]
  __bf16* w_kvb   = (__bf16*)alloc(1048576ull * 2);  // [2048][512]
  __bf16* w_outp  = (__bf16*)alloc(4194304ull * 2);  // [4096][1024]
  float*  qa_f    = (float*)alloc(2097152ull * 4);   // [4096][512]
  __bf16* qa_ln   = (__bf16*)alloc(2097152ull * 2);  // [4096][512]
  __bf16* q_bf    = (__bf16*)alloc(4194304ull * 2);  // [4096][1024]
  float*  ckv_f   = (float*)alloc(512000ull * 4);    // [1000][512]
  __bf16* ckv_bf  = (__bf16*)alloc(512000ull * 2);   // [1000][512]
  __bf16* kv_bf   = (__bf16*)alloc(2048000ull * 2);  // [1000][2048]
  __bf16* k_bf    = (__bf16*)alloc(1048576ull * 2);  // [16][1024][64]
  __bf16* v_bf    = (__bf16*)alloc(1048576ull * 2);  // [16][1024][64]
  __bf16* attn_bf = (__bf16*)alloc(4194304ull * 2);  // [4096][1024]

  auto cvt = [&](const float* in, __bf16* out, long n) {
    long n4 = n >> 2;
    long g = (n4 + 255) / 256;
    if (g > 4096) g = 4096;
    cvt_f32_bf16<<<dim3((unsigned)g), dim3(256), 0, stream>>>(in, out, n4);
  };
  cvt(te, te_bf, 4194304);
  cvt(se, se_bf, 4096000);
  cvt(q_a_w, w_qa, 524288);
  cvt(q_b_w, w_qb, 524288);
  cvt(kv_a_w, w_kva, 2097152);
  cvt(kv_b_w, w_kvb, 1048576);
  cvt(out_w, w_outp, 4194304);

  // q_a = te @ q_a_w^T : [4096,512], f32
  gemm_xwt<false, false><<<dim3(4, 32), dim3(256), 0, stream>>>(
      te_bf, w_qa, nullptr, (void*)qa_f, 4096, 512, 1024);
  ln512_bf16<<<dim3(4096), dim3(256), 0, stream>>>(qa_f, q_ln_g, q_ln_b, qa_ln);
  // q = LN(q_a) @ q_b_w^T : [4096,1024], bf16
  gemm_xwt<true, false><<<dim3(8, 32), dim3(256), 0, stream>>>(
      qa_ln, w_qb, nullptr, (void*)q_bf, 4096, 1024, 512);

  // ckv = se @ kv_a_w^T : [1000,512], f32
  gemm_xwt<false, false><<<dim3(4, 8), dim3(256), 0, stream>>>(
      se_bf, w_kva, nullptr, (void*)ckv_f, 1000, 512, 4096);
  ln512_bf16<<<dim3(1000), dim3(256), 0, stream>>>(ckv_f, kv_ln_g, kv_ln_b, ckv_bf);
  // kv = LN(ckv) @ kv_b_w^T : [1000,2048], bf16
  gemm_xwt<true, false><<<dim3(16, 8), dim3(256), 0, stream>>>(
      ckv_bf, w_kvb, nullptr, (void*)kv_bf, 1000, 2048, 512);

  repack_kv<<<dim3(4096), dim3(256), 0, stream>>>(kv_bf, k_bf, v_bf);

  // attention: grid (B*H, L/128)
  mla_attn<<<dim3(64, 8), dim3(256), 0, stream>>>(q_bf, k_bf, v_bf, attn_bf);

  // final: out = attn @ out_w^T + out_b : [4096,4096], f32 -> d_out
  gemm_xwt<false, true><<<dim3(32, 32), dim3(256), 0, stream>>>(
      attn_bf, w_outp, out_b, d_out, 4096, 4096, 1024);
}
